// SNNModel_23725399343261
// MI455X (gfx1250) — matrix operand order repaired and checked by reading, not compiled, not run on hardware
//
#include <hip/hip_runtime.h>
#include <stdint.h>

#define T_STEPS  1000
#define BATCH    1024
#define IN_F     80
#define INP      96      // K padded to 3x32 for bf16 WMMA
#define HID      128
#define OUTD     10
#define BT       32      // batch rows per workgroup
#define NWAVE    2       // waves per workgroup (16 rows each)
#define NTHREADS 64

typedef __attribute__((ext_vector_type(16))) __bf16 v16bf;
typedef __attribute__((ext_vector_type(8)))  __bf16 v8bf;
typedef __attribute__((ext_vector_type(8)))  float  v8f;

union ABf { v16bf v; v8bf h[2]; };

__device__ __forceinline__ v8f wmma_bf16(v16bf a, v16bf b, v8f c) {
  // D = A(16x32) * B(32x16) + C(16x16 f32)
  return __builtin_amdgcn_wmma_f32_16x16x32_bf16(
      /*neg_a=*/false, a, /*neg_b=*/false, b,
      /*c_mod=*/(short)0, c, /*reuse_a=*/false, /*reuse_b=*/false);
}

__global__ __launch_bounds__(NTHREADS) void snn_forward(
    const float* __restrict__ x, const float* __restrict__ w_in,
    const float* __restrict__ w_rec, const float* __restrict__ w_out,
    float* __restrict__ out)
{
  // Weights as bf16 in LDS, row n contiguous in K so B-fragments are 32B ds loads.
  __shared__ __align__(64) __bf16 s_win[HID][INP];      // 24 KB (K padded, zeros)
  __shared__ __align__(64) __bf16 s_wrec[HID][HID];     // 32 KB
  __shared__ __align__(64) __bf16 s_wout[16][HID];      //  4 KB (rows 10..15 zero)
  __shared__ __align__(64) float  s_x[2][BT][IN_F];     // 20 KB double-buffered x tile
  __shared__ __align__(64) __bf16 s_z[NWAVE][16][HID];  //  8 KB spike relayout bounce
  __shared__ __align__(64) float  s_vo[NWAVE][16][16];  //  2 KB epilogue

  const int tid  = threadIdx.x;
  const int wave = tid >> 5;
  const int lane = tid & 31;
  const int wg   = blockIdx.x;

  // ---- stage weights (f32 -> bf16, zero-pad) ----
  for (int idx = tid; idx < HID * INP; idx += NTHREADS) {
    int n = idx / INP, k = idx - n * INP;
    s_win[n][k] = (k < IN_F) ? (__bf16)w_in[n * IN_F + k] : (__bf16)0.0f;
  }
  for (int idx = tid; idx < HID * HID; idx += NTHREADS)
    s_wrec[idx >> 7][idx & 127] = (__bf16)w_rec[idx];
  for (int idx = tid; idx < 16 * HID; idx += NTHREADS) {
    int n = idx >> 7, k = idx & 127;
    s_wout[n][k] = (n < OUTD) ? (__bf16)w_out[n * HID + k] : (__bf16)0.0f;
  }

  const int m   = lane & 15;          // A: row / B: col / C: N
  const int khi = (lane >> 4) & 1;    // upper lane half
  const int b0  = khi ? 16 : 0;       // A-fragment K offset
  const int kb  = khi ? 16 : 0;       // B-fragment K offset
  const int mhi = khi ? 8 : 0;        // C-layout M offset

  // ---- async prefetch x tile for t=0 (wave-private 16 rows = 5120 B contiguous) ----
  {
    unsigned lds0 = (unsigned)(uintptr_t)(&s_x[0][wave * 16][0]);
    const float* g = x + ((size_t)wg * BT + wave * 16) * IN_F;
#pragma unroll
    for (int j = 0; j < 10; ++j) {
      unsigned off  = (unsigned)(lane * 16 + j * 512);
      unsigned ldsa = lds0 + off;
      asm volatile("global_load_async_to_lds_b128 %0, %1, %2"
                   :: "v"(ldsa), "v"(off), "s"(g) : "memory");
    }
  }

  __syncthreads();  // weights visible

  // ---- persistent per-wave state, WMMA C-layout ----
  v8f vmem[8], icur[8], vo, io;
#pragma unroll
  for (int nt = 0; nt < 8; ++nt)
#pragma unroll
    for (int r = 0; r < 8; ++r) { vmem[nt][r] = 0.0f; icur[nt][r] = 0.0f; }
#pragma unroll
  for (int r = 0; r < 8; ++r) { vo[r] = 0.0f; io[r] = 0.0f; }

  const float*  xrow0 = &s_x[0][wave * 16 + m][0];
  const float*  xrow1 = &s_x[1][wave * 16 + m][0];
  const __bf16* zrow  = &s_z[wave][m][0];
  const __bf16  zone  = (__bf16)1.0f;
  const __bf16  zzero = (__bf16)0.0f;

#pragma unroll 1
  for (int t = 0; t < T_STEPS; ++t) {
    // prefetch t+1 into the other buffer, then wait for current buffer
    if (t + 1 < T_STEPS) {
      unsigned lds0 = (unsigned)(uintptr_t)(&s_x[(t + 1) & 1][wave * 16][0]);
      const float* g = x + ((size_t)(t + 1) * BATCH + (size_t)wg * BT + wave * 16) * IN_F;
#pragma unroll
      for (int j = 0; j < 10; ++j) {
        unsigned off  = (unsigned)(lane * 16 + j * 512);
        unsigned ldsa = lds0 + off;
        asm volatile("global_load_async_to_lds_b128 %0, %1, %2"
                     :: "v"(ldsa), "v"(off), "s"(g) : "memory");
      }
      asm volatile("s_wait_asynccnt 10" ::: "memory");  // in-order: oldest 10 (cur buf) done
    } else {
      asm volatile("s_wait_asynccnt 0" ::: "memory");
    }

    // ---- LIF decay + spike + reset (elementwise, C-layout); spikes -> LDS bf16 ----
#pragma unroll
    for (int nt = 0; nt < 8; ++nt) {
#pragma unroll
      for (int r = 0; r < 8; ++r) {
        float vd  = fmaf(0.1f, icur[nt][r] - vmem[nt][r], vmem[nt][r]); // v + 0.1*(i - v)
        bool  spk = vd > 1.0f;                                          // Heaviside(v_dec - 1)
        vmem[nt][r] = spk ? 0.0f : vd;                                  // reset to 0
        icur[nt][r] = 0.8f * icur[nt][r];                               // i_dec (new C input)
        s_z[wave][r + mhi][nt * 16 + m] = spk ? zone : zzero;
      }
    }

    // ---- input contribution: icur[nt] += x_t @ w_in^T (k-outer, 1 A-frag live) ----
    const float* xr = (t & 1) ? xrow1 : xrow0;
#pragma unroll
    for (int k = 0; k < 3; ++k) {
      ABf a;
#pragma unroll
      for (int e = 0; e < 8; ++e) {
        int K0 = 32 * k + b0 + e;
        int K1 = K0 + 8;
        a.h[0][e] = (K0 < IN_F) ? (__bf16)xr[K0] : (__bf16)0.0f;
        a.h[1][e] = (K1 < IN_F) ? (__bf16)xr[K1] : (__bf16)0.0f;
      }
#pragma unroll
      for (int nt = 0; nt < 8; ++nt) {
        v16bf b = *(const v16bf*)(&s_win[nt * 16 + m][32 * k + kb]);
        icur[nt] = wmma_bf16(a.v, b, icur[nt]);
      }
    }

    // ---- recurrent + readout: icur[nt] += z @ w_rec^T ; jacc += z @ w_out^T ----
    v8f jacc = io;
#pragma unroll
    for (int k = 0; k < 4; ++k) {
      ABf a;
      a.h[0] = *(const v8bf*)(zrow + 32 * k + b0);
      a.h[1] = *(const v8bf*)(zrow + 32 * k + b0 + 8);
#pragma unroll
      for (int nt = 0; nt < 8; ++nt) {
        v16bf b = *(const v16bf*)(&s_wrec[nt * 16 + m][32 * k + kb]);
        icur[nt] = wmma_bf16(a.v, b, icur[nt]);
      }
      v16bf bo = *(const v16bf*)(&s_wout[m][32 * k + kb]);
      jacc = wmma_bf16(a.v, bo, jacc);
    }

    // ---- LI readout decay ----
#pragma unroll
    for (int r = 0; r < 8; ++r) {
      float ij = jacc[r];                 // i_jump = io + z @ w_out^T
      vo[r] = fmaf(0.1f, ij - vo[r], vo[r]);
      io[r] = 0.8f * ij;
    }
  }

  // ---- epilogue: log_softmax over the 10 readout units per batch row ----
#pragma unroll
  for (int r = 0; r < 8; ++r) s_vo[wave][r + mhi][m] = vo[r];
  __syncthreads();
  if (lane < 16) {
    float vals[OUTD];
    float mx = -3.0e38f;
#pragma unroll
    for (int j = 0; j < OUTD; ++j) { vals[j] = s_vo[wave][lane][j]; mx = fmaxf(mx, vals[j]); }
    float s = 0.0f;
#pragma unroll
    for (int j = 0; j < OUTD; ++j) s += __expf(vals[j] - mx);
    float lse = mx + __logf(s);
    float* orow = out + ((size_t)wg * BT + wave * 16 + lane) * OUTD;
#pragma unroll
    for (int j = 0; j < OUTD; ++j) orow[j] = vals[j] - lse;
  }
}

extern "C" void kernel_launch(void* const* d_in, const int* in_sizes, int n_in,
                              void* d_out, int out_size, void* d_ws, size_t ws_size,
                              hipStream_t stream) {
  (void)in_sizes; (void)n_in; (void)d_ws; (void)ws_size; (void)out_size;
  const float* x     = (const float*)d_in[0];
  const float* w_in  = (const float*)d_in[1];
  const float* w_rec = (const float*)d_in[2];
  const float* w_out = (const float*)d_in[3];
  snn_forward<<<BATCH / BT, NTHREADS, 0, stream>>>(
      x, w_in, w_rec, w_out, (float*)d_out);
}
